// ESNLayer_17360257810553
// MI455X (gfx1250) — matrix operand order repaired and checked by reading, not compiled, not run on hardware
//
#include <hip/hip_runtime.h>

// ---------------------------------------------------------------------------
// ESN layer for MI455X (gfx1250).
//   T=512, B=64, I=128, H=1024
//   outputs[t] = h_t = tanh(x_t @ W_ih^T + h_{t-1} @ W_hh^T + b)
//
// Roofline: output traffic 128 MB (~5.5us @ 23.3 TB/s); recurrent chain is
// 512 strictly-sequential [64x1024]x[1024x1024] bf16 WMMA steps -> latency
// bound. Recurrence is independent per batch row, so 4 blocks x 16 batches
// need only __syncthreads() between steps (no grid barrier).
//
//   K1: convert W_ih / W_hh to bf16 in workspace (one pass).
//   K2: parallel WMMA GEMM  U = x @ W_ih^T + b  written into d_out in place.
//   K3: recurrent kernel: hidden state ping-pongs in LDS (bf16); each of 32
//       waves computes two 16x16 tiles (64x v_wmma_f32_16x16x32_bf16 / step);
//       U_t tile (64 KB) is staged global->LDS with GLOBAL_LOAD_ASYNC_TO_LDS
//       issued before the K-loop and waited with s_wait_asynccnt, hiding the
//       global latency behind the WMMA chain.
// ---------------------------------------------------------------------------

#define T_ 512
#define B_ 64
#define I_ 128
#define H_ 1024

typedef __attribute__((ext_vector_type(16))) __bf16 v16bf;
typedef __attribute__((ext_vector_type(8)))  __bf16 v8bf;
typedef __attribute__((ext_vector_type(8)))  float  v8f;
typedef __attribute__((ext_vector_type(4)))  float  v4f;
typedef __attribute__((ext_vector_type(4)))  int    v4i;

#if __has_builtin(__builtin_amdgcn_global_load_async_to_lds_b128) && \
    __has_builtin(__builtin_amdgcn_s_wait_asynccnt)
#define ESN_ASYNC 1
#else
#define ESN_ASYNC 0
#endif

#if ESN_ASYNC
typedef __attribute__((address_space(1))) v4i* gv4i_ptr;
typedef __attribute__((address_space(3))) v4i* lv4i_ptr;
// Generic->AS casts via integer: global generic addrs are the 64-bit VA;
// generic LDS addrs carry the LDS byte offset in addr[31:0] (ISA aperture
// rules), which is exactly the 32-bit AS(3) pointer value.
static __device__ __forceinline__ gv4i_ptr to_global(const void* p) {
  return (gv4i_ptr)(unsigned long long)p;
}
static __device__ __forceinline__ lv4i_ptr to_local(void* p) {
  return (lv4i_ptr)(unsigned int)(unsigned long long)p;
}
#endif

static __device__ __forceinline__ v16bf cat16(v8bf lo, v8bf hi) {
  return __builtin_shufflevector(lo, hi, 0,1,2,3,4,5,6,7,8,9,10,11,12,13,14,15);
}

static __device__ __forceinline__ float fast_tanh(float x) {
  // tanh(|x|) = (1 - e)/(1 + e), e = exp(-2|x|); sign restored. Uses v_exp_f32.
  float ax = fabsf(x);
  float e  = __expf(-2.0f * ax);
  float r  = (1.0f - e) / (1.0f + e);
  return copysignf(r, x);
}

// ---------------------------------------------------------------------------
// K1: f32 -> bf16 weight conversion (W_hh: H*H, W_ih: H*I)
// ---------------------------------------------------------------------------
__global__ void esn_convert(const float* __restrict__ w_ih,
                            const float* __restrict__ w_hh,
                            __bf16* __restrict__ wih_b,
                            __bf16* __restrict__ whh_b) {
  const int total_hh = H_ * H_;   // 1048576
  const int total_ih = H_ * I_;   // 131072
  for (int idx = blockIdx.x * blockDim.x + threadIdx.x; idx < total_hh;
       idx += gridDim.x * blockDim.x) {
    whh_b[idx] = (__bf16)w_hh[idx];
    if (idx < total_ih) wih_b[idx] = (__bf16)w_ih[idx];
  }
}

// ---------------------------------------------------------------------------
// K2: U = x @ W_ih^T + b   -> written into d_out ([T*B, H] f32)
// One wave per 16x16 tile; M = T*B = 32768 (2048 tiles), N = 64 tiles, K = 128.
// ---------------------------------------------------------------------------
__global__ __launch_bounds__(512) void esn_ugemm(const float*  __restrict__ x,
                                                 const __bf16* __restrict__ wih_b,
                                                 const float*  __restrict__ bias,
                                                 float* __restrict__ U) {
  const int wid  = threadIdx.x >> 5;
  const int lane = threadIdx.x & 31;
  const int gw   = blockIdx.x * 16 + wid;   // 131072 waves total
  const int mt   = gw >> 6;                 // 0..2047
  const int nt   = gw & 63;                 // 0..63
  const int row  = lane & 15;
  const int kh   = lane >> 4;               // K-half select per ISA layout

  const float*  arow = x + (size_t)(mt * 16 + row) * I_;
  const __bf16* brow = wih_b + (size_t)(nt * 16 + row) * I_ + kh * 8;

  v8f c = {};
#pragma unroll
  for (int ks = 0; ks < 4; ++ks) {
    const int k0 = ks * 32;
    // A operand, 16-bit A 16x32 layout: lane<16 holds K {0..7,16..23},
    // lane>=16 holds K {8..15,24..31} of this 32-block (row = lane&15).
    v4f a0 = *(const v4f*)(arow + k0 + kh * 8);
    v4f a1 = *(const v4f*)(arow + k0 + kh * 8 + 4);
    v4f a2 = *(const v4f*)(arow + k0 + 16 + kh * 8);
    v4f a3 = *(const v4f*)(arow + k0 + 16 + kh * 8 + 4);
    v16bf av;
#pragma unroll
    for (int e = 0; e < 4; ++e) {
      av[e]      = (__bf16)a0[e];
      av[e + 4]  = (__bf16)a1[e];
      av[e + 8]  = (__bf16)a2[e];
      av[e + 12] = (__bf16)a3[e];
    }
    // B operand, 32x16 layout: lane holds column n = lane&15, same K order as A.
    v8bf b0 = *(const v8bf*)(brow + k0);
    v8bf b1 = *(const v8bf*)(brow + k0 + 16);
    c = __builtin_amdgcn_wmma_f32_16x16x32_bf16(false, av, false, cat16(b0, b1),
                                                (short)0, c, false, false);
  }

  const int n  = nt * 16 + row;
  const float bv = bias[n];
  float* out = U + (size_t)(mt * 16) * H_ + n;
#pragma unroll
  for (int j = 0; j < 8; ++j) {
    const int m = j + kh * 8;            // C/D layout: vgpr j, lanes>=16 -> m+8
    out[(size_t)m * H_] = c[j] + bv;
  }
}

// ---------------------------------------------------------------------------
// K3: recurrent chain. 4 blocks x 1024 threads; block bb owns batches
// [bb*16, bb*16+16). Hidden state (16x1024 bf16) ping-pongs in LDS; U_t tile
// staged into LDS asynchronously while the WMMA K-loop runs.
// ---------------------------------------------------------------------------
__global__ __launch_bounds__(1024) void esn_recur(const float*  __restrict__ h0,
                                                  const __bf16* __restrict__ whh_b,
                                                  float* __restrict__ out,      // [T,B,H]; U in, h out (in place)
                                                  float* __restrict__ hfinal) { // [B,H]
  __shared__ __align__(16) __bf16 hbuf[2][16 * H_];   // 64 KB
#if ESN_ASYNC
  __shared__ __align__(16) float  ubuf[16 * H_];      // 64 KB U staging
#endif

  const int bb   = blockIdx.x;          // batch block 0..3
  const int tid  = threadIdx.x;
  const int wid  = tid >> 5;            // 0..31
  const int lane = tid & 31;
  const int row  = lane & 15;
  const int kh   = lane >> 4;

  // Stage h0 (f32) into LDS bf16.
  for (int idx = tid; idx < 16 * H_; idx += 1024) {
    const int m = idx >> 10;
    const int k = idx & (H_ - 1);
    hbuf[0][idx] = (__bf16)h0[(size_t)(bb * 16 + m) * H_ + k];
  }
  __syncthreads();

  const int nt0 = wid * 2, nt1 = wid * 2 + 1;
  const __bf16* b0row = whh_b + (size_t)(nt0 * 16 + row) * H_ + kh * 8;
  const __bf16* b1row = whh_b + (size_t)(nt1 * 16 + row) * H_ + kh * 8;
  const int n0 = nt0 * 16 + row;
  const int n1 = nt1 * 16 + row;

  for (int t = 0; t < T_; ++t) {
    float* Ut = out + (size_t)t * (B_ * H_) + (size_t)(bb * 16) * H_;

#if ESN_ASYNC
    // Kick off async global->LDS DMA of this step's 64 KB U tile; it overlaps
    // with the 64-WMMA K-loop below. 1024 threads x 4 x b128 = 64 KB.
    {
      const char* g = (const char*)Ut;
      char*       l = (char*)&ubuf[0];
      const int   o = tid << 4;           // 16B per thread per chunk
#pragma unroll
      for (int i = 0; i < 4; ++i) {
        __builtin_amdgcn_global_load_async_to_lds_b128(
            to_global(g + o + i * 16384), to_local(l + o + i * 16384), 0, 0);
      }
    }
#else
    if (t + 1 < T_) {
      const float* nxt = out + (size_t)(t + 1) * (B_ * H_) + (size_t)(bb * 16) * H_;
      __builtin_prefetch(nxt + (tid << 4), 0, 0);
    }
#endif

    const __bf16* abase = &hbuf[t & 1][0] + row * H_ + kh * 8;
    v8f c0 = {}, c1 = {};
#pragma unroll 4
    for (int ks = 0; ks < 32; ++ks) {
      const int k0 = ks * 32;
      // A from LDS (ds_load_b128), shared by all 32 waves in the block.
      v8bf alo = *(const v8bf*)(abase + k0);
      v8bf ahi = *(const v8bf*)(abase + k0 + 16);
      v16bf av = cat16(alo, ahi);
      // B tiles from L2-resident bf16 W_hh.
      v8bf bl0 = *(const v8bf*)(b0row + k0);
      v8bf bh0 = *(const v8bf*)(b0row + k0 + 16);
      c0 = __builtin_amdgcn_wmma_f32_16x16x32_bf16(false, av, false, cat16(bl0, bh0),
                                                   (short)0, c0, false, false);
      v8bf bl1 = *(const v8bf*)(b1row + k0);
      v8bf bh1 = *(const v8bf*)(b1row + k0 + 16);
      c1 = __builtin_amdgcn_wmma_f32_16x16x32_bf16(false, av, false, cat16(bl1, bh1),
                                                   (short)0, c1, false, false);
    }

#if ESN_ASYNC
    // Own async ops done, then block-wide barrier so every wave sees the
    // whole staged tile.
    __builtin_amdgcn_s_wait_asynccnt(0);
    __syncthreads();
    const float* Us = &ubuf[0];
#else
    const float* Us = Ut;
#endif

    __bf16* hnext = &hbuf[(t + 1) & 1][0];
#pragma unroll
    for (int j = 0; j < 8; ++j) {
      const int m = j + kh * 8;
      const size_t r = (size_t)m * H_;
      const float v0 = fast_tanh(Us[r + n0] + c0[j]);
      const float v1 = fast_tanh(Us[r + n1] + c1[j]);
      Ut[r + n0] = v0;
      Ut[r + n1] = v1;
      hnext[m * H_ + n0] = (__bf16)v0;
      hnext[m * H_ + n1] = (__bf16)v1;
      if (t == T_ - 1) {
        hfinal[(size_t)(bb * 16 + m) * H_ + n0] = v0;
        hfinal[(size_t)(bb * 16 + m) * H_ + n1] = v1;
      }
    }
    __syncthreads();
  }
}

// ---------------------------------------------------------------------------
// Host-side launcher
// Inputs (setup_inputs order): x[T,B,I] f32, h0[B,H] f32, w_ih[H,I] f32,
//                              w_hh[H,H] f32, b[H] f32
// d_out: outputs[T,B,H] f32 then h_final[B,H] f32, concatenated flat.
// ---------------------------------------------------------------------------
extern "C" void kernel_launch(void* const* d_in, const int* in_sizes, int n_in,
                              void* d_out, int out_size, void* d_ws, size_t ws_size,
                              hipStream_t stream) {
  (void)in_sizes; (void)n_in; (void)out_size; (void)ws_size;

  const float* x    = (const float*)d_in[0];
  const float* h0   = (const float*)d_in[1];
  const float* wih  = (const float*)d_in[2];
  const float* whh  = (const float*)d_in[3];
  const float* bias = (const float*)d_in[4];

  float* out = (float*)d_out;

  char* ws = (char*)d_ws;
  __bf16* whh_b = (__bf16*)ws;                             // 2 MB
  __bf16* wih_b = (__bf16*)(ws + (size_t)2 * 1024 * 1024); // 256 KB

  // K1: weight conversion (exactly one element per thread for W_hh).
  esn_convert<<<4096, 256, 0, stream>>>(wih, whh, wih_b, whh_b);

  // K2: U = x @ W_ih^T + b  (131072 tiles / 16 waves per block).
  esn_ugemm<<<8192, 512, 0, stream>>>(x, wih_b, bias, out);

  // K3: sequential recurrence; 4 blocks own 16 batches each.
  esn_recur<<<4, 1024, 0, stream>>>(h0, whh_b, out,
                                    out + (size_t)T_ * B_ * H_);
}
